// MultiHeadAttention_73864847556656
// MI455X (gfx1250) — compile-verified, matrix operand order checked by reading, not run on hardware
//
#include <hip/hip_runtime.h>
#include <hip/hip_bf16.h>

#define NH     8
#define DMODEL 512
#define DEPTH  64
#define SEQ    2048
#define BATCH  4

typedef __attribute__((ext_vector_type(16))) __bf16 bf16x16;
typedef __attribute__((ext_vector_type(8)))  __bf16 bf16x8;
typedef __attribute__((ext_vector_type(8)))  float  floatx8;
typedef __attribute__((ext_vector_type(4)))  float  f32x4;   // native ext-vector (NT-builtin OK)

__device__ __forceinline__ int lane_id() { return threadIdx.x & 31; }

// ---- Fragment loaders (ISA 7.12.2 16-bit A/B^T layout) ----
// Per lane: elems 0..7  <-> k = g*8 + 0..7   (g = lane>>4)
//           elems 8..15 <-> k = 16 + g*8 + 0..7
// Both runs are 8 contiguous elements -> b128 vector loads.

__device__ __forceinline__ bf16x16 frag_bf16(const __bf16* __restrict__ base, int ld) {
    int L = lane_id(), r = L & 15, g = L >> 4;
    const __bf16* p = base + (size_t)r * ld + g * 8;
    bf16x8 lo = *(const bf16x8*)(p);
    bf16x8 hi = *(const bf16x8*)(p + 16);
    return __builtin_shufflevector(lo, hi, 0,1,2,3,4,5,6,7,8,9,10,11,12,13,14,15);
}

__device__ __forceinline__ bf16x16 pack_frag(f32x4 x0, f32x4 x1, f32x4 x2, f32x4 x3) {
    bf16x16 f;
    f[0]=(__bf16)x0[0];  f[1]=(__bf16)x0[1];  f[2]=(__bf16)x0[2];  f[3]=(__bf16)x0[3];
    f[4]=(__bf16)x1[0];  f[5]=(__bf16)x1[1];  f[6]=(__bf16)x1[2];  f[7]=(__bf16)x1[3];
    f[8]=(__bf16)x2[0];  f[9]=(__bf16)x2[1];  f[10]=(__bf16)x2[2]; f[11]=(__bf16)x2[3];
    f[12]=(__bf16)x3[0]; f[13]=(__bf16)x3[1]; f[14]=(__bf16)x3[2]; f[15]=(__bf16)x3[3];
    return f;
}

// fp32 source, k contiguous (b128 loads + pack-convert)
__device__ __forceinline__ bf16x16 frag_f32(const float* __restrict__ base, int ld) {
    int L = lane_id(), r = L & 15, g = L >> 4;
    const float* p = base + (size_t)r * ld + g * 8;
    return pack_frag(*(const f32x4*)(p),      *(const f32x4*)(p + 4),
                     *(const f32x4*)(p + 16), *(const f32x4*)(p + 20));
}

// same, non-temporal (for the streaming 536 MB attn tensor: > L2, read once)
__device__ __forceinline__ bf16x16 frag_f32_nt(const float* __restrict__ base, int ld) {
    int L = lane_id(), r = L & 15, g = L >> 4;
    const float* p = base + (size_t)r * ld + g * 8;
    f32x4 x0 = __builtin_nontemporal_load((const f32x4*)(p));
    f32x4 x1 = __builtin_nontemporal_load((const f32x4*)(p + 4));
    f32x4 x2 = __builtin_nontemporal_load((const f32x4*)(p + 16));
    f32x4 x3 = __builtin_nontemporal_load((const f32x4*)(p + 20));
    return pack_frag(x0, x1, x2, x3);
}

__device__ __forceinline__ floatx8 wmma_bf16(bf16x16 a, bf16x16 b, floatx8 c) {
    return __builtin_amdgcn_wmma_f32_16x16x32_bf16(false, a, false, b,
                                                   (short)0, c, false, false);
}

// ---------------- Kernel 0: W (K x N fp32) -> W^T (N x K bf16)
__global__ __launch_bounds__(256)
void wconv_kernel(const float* __restrict__ W, __bf16* __restrict__ T) {
    int e  = blockIdx.x * blockDim.x + threadIdx.x;    // 512*512 threads
    int n  = e >> 9, kk = e & (DMODEL - 1);            // consecutive threads vary kk -> coalesced store
    T[(size_t)n * DMODEL + kk] = (__bf16)W[(size_t)kk * DMODEL + n];
}

// ---------------- Kernel 1: one projection -> bf16 split-head
// TRANSPOSE_V: store v transposed within head: vt[bh][d][t] (k-contig B for ctx GEMM)
template <bool TRANSPOSE_V>
__global__ __launch_bounds__(256)
void proj_kernel(const float* __restrict__ A, const __bf16* __restrict__ WT,
                 const float* __restrict__ bias, __bf16* __restrict__ dst) {
    int wave = (blockIdx.x * blockDim.x + threadIdx.x) >> 5;
    const int colTiles = DMODEL / 16;                  // 32
    int rt = wave / colTiles, ct = wave % colTiles;

    floatx8 acc = {};
    for (int kc = 0; kc < DMODEL; kc += 32) {
        bf16x16 af = frag_f32 (A  + (size_t)(rt * 16) * DMODEL + kc, DMODEL);
        bf16x16 bf = frag_bf16(WT + (size_t)(ct * 16) * DMODEL + kc, DMODEL);
        acc = wmma_bf16(af, bf, acc);
    }

    int L = lane_id(), n = L & 15, g = L >> 4;
    int col = ct * 16 + n;
    int h = col / DEPTH, dd = col % DEPTH;
    float bval = bias[col];
#pragma unroll
    for (int r2 = 0; r2 < 8; ++r2) {
        int m = rt * 16 + r2 + 8 * g;                  // global row over B*S
        int b = m / SEQ, s = m % SEQ;
        __bf16 val = (__bf16)(acc[r2] + bval);
        if constexpr (TRANSPOSE_V)
            dst[(((size_t)b * NH + h) * DEPTH + dd) * SEQ + s] = val;
        else
            dst[(((size_t)b * NH + h) * SEQ + s) * DEPTH + dd] = val;
    }
}

// ---------------- Kernel 2: logits (QK^T + Q*pos^T)/8 -> LDS, softmax, write attn
__global__ __launch_bounds__(256)
void attn_kernel(const __bf16* __restrict__ qh, const __bf16* __restrict__ kh,
                 const __bf16* __restrict__ ph, float* __restrict__ attn) {
    extern __shared__ float lds[];                 // 16 rows x SEQ fp32 = 128 KB
    int bh   = blockIdx.y;                         // b*NH + h
    int st   = blockIdx.x;                         // 16-row stripe
    int wave = threadIdx.x >> 5;
    int L    = lane_id();

    const __bf16* qb = qh + ((size_t)bh * SEQ + st * 16) * DEPTH;
    bf16x16 aq0 = frag_bf16(qb,      DEPTH);       // k = 0..31
    bf16x16 aq1 = frag_bf16(qb + 32, DEPTH);       // k = 32..63
    const __bf16* kb0 = kh + (size_t)bh * SEQ * DEPTH;
    const __bf16* pb0 = ph + (size_t)bh * SEQ * DEPTH;

    for (int nt = wave; nt < SEQ / 16; nt += 8) {
        const __bf16* kb = kb0 + (size_t)nt * 16 * DEPTH;
        const __bf16* pb = pb0 + (size_t)nt * 16 * DEPTH;
        bf16x16 bk0 = frag_bf16(kb,      DEPTH);   // kh acts as B^T
        bf16x16 bk1 = frag_bf16(kb + 32, DEPTH);
        bf16x16 bp0 = frag_bf16(pb,      DEPTH);
        bf16x16 bp1 = frag_bf16(pb + 32, DEPTH);
        floatx8 acc = {};
        acc = wmma_bf16(aq0, bk0, acc);
        acc = wmma_bf16(aq1, bk1, acc);
        acc = wmma_bf16(aq0, bp0, acc);
        acc = wmma_bf16(aq1, bp1, acc);
        int n = L & 15, g = L >> 4;
#pragma unroll
        for (int r2 = 0; r2 < 8; ++r2)
            lds[(size_t)(r2 + 8 * g) * SEQ + nt * 16 + n] = acc[r2] * 0.125f;
    }
    __syncthreads();

    // softmax: each wave owns rows 2w, 2w+1 (wave32 shuffle reductions)
#pragma unroll
    for (int rr = 0; rr < 2; ++rr) {
        int row = wave * 2 + rr;
        float* rp = lds + (size_t)row * SEQ;
        float mx = -3.0e38f;
        for (int c = L; c < SEQ; c += 32) mx = fmaxf(mx, rp[c]);
#pragma unroll
        for (int off = 16; off > 0; off >>= 1) mx = fmaxf(mx, __shfl_xor(mx, off, 32));
        float sum = 0.f;
        for (int c = L; c < SEQ; c += 32) {
            float e = __expf(rp[c] - mx);
            rp[c] = e;
            sum += e;
        }
#pragma unroll
        for (int off = 16; off > 0; off >>= 1) sum += __shfl_xor(sum, off, 32);
        float inv = 1.f / sum;
        float* op = attn + ((size_t)bh * SEQ + st * 16 + row) * SEQ;
        for (int c = L * 4; c < SEQ; c += 128) {
            f32x4 vv;
            vv[0] = rp[c] * inv;     vv[1] = rp[c + 1] * inv;
            vv[2] = rp[c + 2] * inv; vv[3] = rp[c + 3] * inv;
            __builtin_nontemporal_store(vv, (f32x4*)(op + c));   // NT b128: don't thrash L2
        }
    }
}

// ---------------- Kernel 3: ctx = attn @ v -- wave owns a full 16x64 stripe,
// so the 536 MB attn tensor is streamed exactly once (NT loads).
__global__ __launch_bounds__(256)
void ctx_kernel(const float* __restrict__ attn, const __bf16* __restrict__ vt,
                __bf16* __restrict__ ctx) {
    int wave = (blockIdx.x * blockDim.x + threadIdx.x) >> 5;
    const int rowTiles = SEQ / 16;                 // 128
    int bh = wave / rowTiles;
    int rt = wave % rowTiles;

    const float*  ab  = attn + ((size_t)bh * SEQ + rt * 16) * SEQ;
    const __bf16* vtb = vt   + (size_t)bh * DEPTH * SEQ;   // [d][t], k-contig over t
    floatx8 acc0 = {}, acc1 = {}, acc2 = {}, acc3 = {};
    for (int kc = 0; kc < SEQ; kc += 32) {
        bf16x16 af = frag_f32_nt(ab + kc, SEQ);
        acc0 = wmma_bf16(af, frag_bf16(vtb +  0 * SEQ + kc, SEQ), acc0);
        acc1 = wmma_bf16(af, frag_bf16(vtb + 16 * SEQ + kc, SEQ), acc1);
        acc2 = wmma_bf16(af, frag_bf16(vtb + 32 * SEQ + kc, SEQ), acc2);
        acc3 = wmma_bf16(af, frag_bf16(vtb + 48 * SEQ + kc, SEQ), acc3);
    }
    int L = lane_id(), n = L & 15, g = L >> 4;
    __bf16* db = ctx + ((size_t)bh * SEQ + rt * 16) * DEPTH;
#pragma unroll
    for (int r2 = 0; r2 < 8; ++r2) {
        size_t row = (size_t)(r2 + 8 * g) * DEPTH;
        db[row +  0 + n] = (__bf16)acc0[r2];
        db[row + 16 + n] = (__bf16)acc1[r2];
        db[row + 32 + n] = (__bf16)acc2[r2];
        db[row + 48 + n] = (__bf16)acc3[r2];
    }
}

// ---------------- Kernel 4: out = merge_heads(ctx) @ Wd + bd
__global__ __launch_bounds__(256)
void outproj_kernel(const __bf16* __restrict__ ctx, const __bf16* __restrict__ WdT,
                    const float* __restrict__ bd, float* __restrict__ out) {
    int wave = (blockIdx.x * blockDim.x + threadIdx.x) >> 5;
    const int colTiles = DMODEL / 16;              // 32
    int rt = wave / colTiles, ct = wave % colTiles;
    int L = lane_id(), r = L & 15, g = L >> 4;
    int R = rt * 16 + r;                           // global row over B*S
    int b = R / SEQ, s = R % SEQ;
    const __bf16* cb = ctx + ((size_t)b * NH * SEQ + s) * DEPTH;

    floatx8 acc = {};
    for (int kc = 0; kc < DMODEL; kc += 32) {
        // gather A across heads: run of 8 stays inside one 64-wide head block
        int k1 = kc + g * 8, k2 = k1 + 16;
        const __bf16* p1 = cb + (size_t)(k1 / DEPTH) * SEQ * DEPTH + (k1 % DEPTH);
        const __bf16* p2 = cb + (size_t)(k2 / DEPTH) * SEQ * DEPTH + (k2 % DEPTH);
        bf16x8 lo = *(const bf16x8*)p1;
        bf16x8 hi = *(const bf16x8*)p2;
        bf16x16 af = __builtin_shufflevector(lo, hi, 0,1,2,3,4,5,6,7,8,9,10,11,12,13,14,15);
        bf16x16 bf = frag_bf16(WdT + (size_t)(ct * 16) * DMODEL + kc, DMODEL);
        acc = wmma_bf16(af, bf, acc);
    }
    int n = L & 15;
    int col = ct * 16 + n;
    float bval = bd[col];
    float* ob = out + (size_t)(rt * 16) * DMODEL + col;
#pragma unroll
    for (int r2 = 0; r2 < 8; ++r2)
        ob[(size_t)(r2 + 8 * g) * DMODEL] = acc[r2] + bval;
}

extern "C" void kernel_launch(void* const* d_in, const int* in_sizes, int n_in,
                              void* d_out, int out_size, void* d_ws, size_t ws_size,
                              hipStream_t stream) {
    (void)in_sizes; (void)n_in; (void)out_size; (void)ws_size;
    const float* q  = (const float*)d_in[0];
    const float* k  = (const float*)d_in[1];
    const float* v  = (const float*)d_in[2];
    const float* pe = (const float*)d_in[3];
    /* d_in[4] = mask (unused by reference) */
    const float* Wq = (const float*)d_in[5];
    const float* bq = (const float*)d_in[6];
    const float* Wk = (const float*)d_in[7];
    const float* bk = (const float*)d_in[8];
    const float* Wv = (const float*)d_in[9];
    const float* bv = (const float*)d_in[10];
    const float* Wd = (const float*)d_in[11];
    const float* bd = (const float*)d_in[12];

    float* out  = (float*)d_out;
    float* attn = out + (size_t)BATCH * SEQ * DMODEL;      // outputs concat: (out, attn)

    // workspace (all bf16): 4 transposed weights + qh/kh/vt/ph + ctx  (~36 MB; ctx aliases qh)
    const size_t WSZ = (size_t)DMODEL * DMODEL;            // 262144
    const size_t PER = (size_t)BATCH * SEQ * DMODEL;       // 4194304
    __bf16* wsb = (__bf16*)d_ws;
    __bf16* WqT = wsb;
    __bf16* WkT = WqT + WSZ;
    __bf16* WvT = WkT + WSZ;
    __bf16* WdT = WvT + WSZ;
    __bf16* qh  = WdT + WSZ;
    __bf16* kh  = qh + PER;
    __bf16* vt  = kh + PER;
    __bf16* ph  = vt + PER;
    __bf16* ctx = qh;                                      // qh dead after attn_kernel

    {   // weight convert + transpose (4 small launches, uniform code)
        int blocks = (int)WSZ / 256;
        wconv_kernel<<<blocks, 256, 0, stream>>>(Wq, WqT);
        wconv_kernel<<<blocks, 256, 0, stream>>>(Wk, WkT);
        wconv_kernel<<<blocks, 256, 0, stream>>>(Wv, WvT);
        wconv_kernel<<<blocks, 256, 0, stream>>>(Wd, WdT);
    }
    {   // projections: 4 launches, straight-line code (no per-wave matrix select)
        int waves  = ((BATCH * SEQ) / 16) * (DMODEL / 16); // 16384
        int blocks = waves * 32 / 256;                     // 2048
        proj_kernel<false><<<blocks, 256, 0, stream>>>(q,  WqT, bq, qh);
        proj_kernel<false><<<blocks, 256, 0, stream>>>(k,  WkT, bk, kh);
        proj_kernel<true ><<<blocks, 256, 0, stream>>>(v,  WvT, bv, vt);
        proj_kernel<false><<<blocks, 256, 0, stream>>>(pe, WdT, bd, ph);
    }
    {   // scores + softmax + attn output (128 KB dynamic LDS per block)
        size_t sh = (size_t)16 * SEQ * sizeof(float);
        (void)hipFuncSetAttribute((const void*)attn_kernel,
                                  hipFuncAttributeMaxDynamicSharedMemorySize, (int)sh);
        dim3 grid(SEQ / 16, BATCH * NH);
        attn_kernel<<<grid, 256, sh, stream>>>(qh, kh, ph, attn);
    }
    {   // ctx = attn @ v (one wave per 16-row stripe, all 4 depth tiles)
        int waves = BATCH * NH * (SEQ / 16);
        ctx_kernel<<<waves * 32 / 256, 256, 0, stream>>>(attn, vt, ctx);
    }
    {   // output projection
        int waves = ((BATCH * SEQ) / 16) * (DMODEL / 16);
        outproj_kernel<<<waves * 32 / 256, 256, 0, stream>>>(ctx, WdT, bd, out);
    }
}
